// HawkesCTLSTM_6373731467910
// MI455X (gfx1250) — matrix operand order repaired and checked by reading, not compiled, not run on hardware
//
#include <hip/hip_runtime.h>

typedef __attribute__((ext_vector_type(16))) _Float16 v16h;
typedef __attribute__((ext_vector_type(8)))  float    v8f;

#define P_DIM  16
#define H_DIM  128
#define L_SEQ  1024
#define B_SZ   256
#define NTYPES 17

// ---- dynamic LDS layout (in f16 halves) ----
// Per-(gate,ntile) fragment BLOCK: 4 h-Ktile frags + 1 embed frag, contiguous.
//   block = 5 * 512 halves = 2560 halves (5120 B)  -> all loads via DS imm offsets
// 48 gate blocks (g 0..5 x nt 0..7) + 1 decay block.
#define BLK_HALFS 2560
#define FRAG0   0
#define DECBLK  48                         // decay block index
#define ONEH0   (49*BLK_HALFS)             // 125440: one-hot A table, 17*32 halves
#define HBASE0  (ONEH0 + 544)              // 125984: h ping-pong, 2 x 2048 halves
#define H_BUF_HALFS (16*H_DIM)             // 2048
#define SMEM_BYTES ((HBASE0 + 2*H_BUF_HALFS) * 2)   // 260160 B  (< 320KB WGP LDS)

__device__ __forceinline__ float fast_sigm(float x) {
    return __builtin_amdgcn_rcpf(1.0f + __expf(-x));
}
__device__ __forceinline__ float fast_tanh(float x) {
    // tanh(x) = 2*sigmoid(2x) - 1   (one v_exp + one v_rcp)
    return 2.0f * __builtin_amdgcn_rcpf(1.0f + __expf(-2.0f * x)) - 1.0f;
}

__device__ __forceinline__ v16h ld_frag(const _Float16* p) {
    v16h v;
    ((uint4*)&v)[0] = *(const uint4*)p;
    ((uint4*)&v)[1] = *(const uint4*)(p + 8);
    return v;
}

__global__ __launch_bounds__(256, 1)
void ctlstm_kernel(const float* __restrict__ dt, const int* __restrict__ types,
                   const float* __restrict__ h0, const float* __restrict__ c0,
                   const float* __restrict__ embed,
                   const float* Wi, const float* bi, const float* Wf, const float* bf,
                   const float* Wo, const float* bo, const float* Wit, const float* bit_,
                   const float* Wft, const float* bft, const float* Wz, const float* bz,
                   const float* Wd, const float* bd,
                   float* __restrict__ out)
{
    extern __shared__ char smem[];
    _Float16* lds = (_Float16*)smem;

    const float* Wg[6] = {Wi, Wf, Wo, Wit, Wft, Wz};
    const float* bg[6] = {bi, bf, bo, bit_, bft, bz};

    const int tid  = threadIdx.x;
    const int lane = tid & 31;
    const int wv   = tid >> 5;         // wave id 0..7 -> hidden n-tile
    const int b0   = blockIdx.x * 16;  // batch tile base
    const int hi   = lane >> 4;        // lane half (0/1)
    const int lm   = lane & 15;        // 0..15

    // ---------- one-time setup (per workgroup; trivial vs. 1024-step loop) ----------
    // (a) h-part gate weight fragments into per-(g,nt) blocks.
    //   element e of lane ln -> W[k][n], n = nt*16+(ln&15), k = kt*32+(ln>>4)*8+(e>>3)*16+(e&7)
    for (int i = tid; i < 6*8*4*512; i += 256) {
        int e  = i & 15;
        int ln = (i >> 4) & 31;
        int kt = (i >> 9) & 3;
        int nt = (i >> 11) & 7;
        int g  = i >> 14;
        int n  = nt * 16 + (ln & 15);
        int k  = kt * 32 + (ln >> 4) * 8 + (e >> 3) * 16 + (e & 7);
        lds[(g * 8 + nt) * BLK_HALFS + kt * 512 + ln * 16 + e] =
            (_Float16)Wg[g][(P_DIM + k) * H_DIM + n];
    }
    // (b) embed-part fragments G' = [embed@W (16 rows); bias; zeros], slot 4 of each block
    for (int i = tid; i < 48*512; i += 256) {
        int e  = i & 15;
        int ln = (i >> 4) & 31;
        int nt = (i >> 9) & 7;
        int g  = i >> 12;
        int n  = nt * 16 + (ln & 15);
        int k  = (ln >> 4) * 8 + (e >> 3) * 16 + (e & 7);   // 0..31 within tile
        float v = 0.0f;
        if (k < P_DIM) {
            #pragma unroll
            for (int p = 0; p < P_DIM; ++p) v += embed[k * P_DIM + p] * Wg[g][p * H_DIM + n];
        } else if (k == P_DIM) {
            v = bg[g][n];
        }
        lds[(g * 8 + nt) * BLK_HALFS + 4 * 512 + ln * 16 + e] = (_Float16)v;
    }
    // (c) decay block: 4 h-Ktile frags (col 0 = Wd[16:144]) + embed frag (col 0 = [embed@Wd; bd; 0])
    for (int i = tid; i < 5*512; i += 256) {
        int e  = i & 15;
        int ln = (i >> 4) & 31;
        int kt = i >> 9;                 // 0..4 (4 = embed slot)
        int k  = (kt < 4 ? kt * 32 : 0) + (ln >> 4) * 8 + (e >> 3) * 16 + (e & 7);
        float v = 0.0f;
        if ((ln & 15) == 0) {
            if (kt < 4) {
                v = Wd[P_DIM + k];
            } else if (k < P_DIM) {
                #pragma unroll
                for (int p = 0; p < P_DIM; ++p) v += embed[k * P_DIM + p] * Wd[p];
            } else if (k == P_DIM) {
                v = bd[0];
            }
        }
        lds[DECBLK * BLK_HALFS + kt * 512 + ln * 16 + e] = (_Float16)v;
    }
    // (d) one-hot A-fragment table: [type][half][e] -> 1.0 if k==type or k==16 (bias row)
    for (int i = tid; i < NTYPES*32; i += 256) {
        int e  = i & 15;
        int h2 = (i >> 4) & 1;
        int ty = i >> 5;
        int k  = h2 * 8 + (e >> 3) * 16 + (e & 7);
        lds[ONEH0 + i] = (k == ty || k == P_DIM) ? (_Float16)1.0f : (_Float16)0.0f;
    }
    // (e) h state (f16, row-major), buffer 0 = step 0 input
    for (int i = tid; i < 16 * H_DIM; i += 256) {
        int m = i >> 7, n = i & 127;
        lds[HBASE0 + i] = (_Float16)h0[(size_t)(b0 + m) * H_DIM + n];
    }

    // Per-lane recurrent state: C/D-layout element (r,lane) = (m = r+8*hi, n = wv*16+lm)
    float cc[8], ctg[8];
    #pragma unroll
    for (int r = 0; r < 8; ++r) {
        size_t ix = (size_t)(b0 + r + 8 * hi) * H_DIM + wv * 16 + lm;
        cc[r]  = c0[ix];
        ctg[r] = c0[ix];
    }

    // Hoisted, loop-invariant per-gate fragment base pointers (imm offsets cover a block)
    const _Float16* bp[7];
    #pragma unroll
    for (int g = 0; g < 6; ++g) bp[g] = lds + (g * 8 + wv) * BLK_HALFS + lane * 16;
    bp[6] = lds + DECBLK * BLK_HALFS + lane * 16;

    __syncthreads();

    const size_t LBH = (size_t)L_SEQ * B_SZ * H_DIM;

    #pragma unroll 1
    for (int t = 0; t < L_SEQ; ++t) {
        const _Float16* hcur = lds + HBASE0 + (t & 1) * H_BUF_HALFS;
        _Float16*       hnxt = lds + HBASE0 + ((t + 1) & 1) * H_BUF_HALFS;

        // per-row scalars for row lm (lanes 16..31 mirror 0..15)
        float dt_m = dt[t * B_SZ + b0 + lm];
        int   ty_m = types[t * B_SZ + b0 + lm];
        if (t + 1 < L_SEQ) {
            __builtin_prefetch(dt + (t + 1) * B_SZ + b0, 0, 0);
            __builtin_prefetch(types + (t + 1) * B_SZ + b0, 0, 0);
        }

        // A fragments: 4 K-tiles of h + 1 one-hot(type)/bias tile
        v16h af[5];
        #pragma unroll
        for (int kt = 0; kt < 4; ++kt) {
            const _Float16* ap = hcur + lm * H_DIM + kt * 32 + hi * 8;
            ((uint4*)&af[kt])[0] = *(const uint4*)ap;
            ((uint4*)&af[kt])[1] = *(const uint4*)(ap + 16);
        }
        af[4] = ld_frag(lds + ONEH0 + (ty_m * 2 + hi) * 16);

        // 35 WMMAs, K-tile-major; gate order rotated per K-tile so each accumulator's
        // chain stays ~7 WMMAs apart (no D->C hazard NOPs).
        v8f acc[7];
        #pragma unroll
        for (int g = 0; g < 7; ++g) acc[g] = (v8f){};

        #pragma unroll
        for (int kt = 0; kt < 5; ++kt) {
            #pragma unroll
            for (int gg = 0; gg < 7; ++gg) {
                int g = (gg + kt) % 7;
                v16h bfr = ld_frag(bp[g] + kt * 512);   // imm offset: kt*1024 B (+16)
                acc[g] = __builtin_amdgcn_wmma_f32_16x16x32_f16(
                    false, af[kt], false, bfr, (short)0, acc[g], false, false);
            }
        }

        // decay: column 0 of acc[6]. Lane 0 holds rows 0..7, lane 16 rows 8..15.
        // All lanes run the math (only lanes 0/16 produce meaningful values).
        float dtr[8], dec[8], edt[8];
        #pragma unroll
        for (int r = 0; r < 8; ++r) dtr[r] = __shfl(dt_m, r + 8 * hi, 32);
        #pragma unroll
        for (int r = 0; r < 8; ++r) {
            float dpre = acc[6][r];
            float x8   = 8.0f * dpre;
            dec[r] = (x8 > 20.0f) ? dpre : 0.125f * __logf(1.0f + __expf(x8));
            edt[r] = __expf(-dec[r] * dtr[r]);
        }
        // broadcast exp(-decay*dt) from lane 0 (rows 0..7) / lane 16 (rows 8..15)
        float edtr[8];
        #pragma unroll
        for (int r = 0; r < 8; ++r) edtr[r] = __shfl(edt[r], hi << 4, 32);

        // elementwise CT-LSTM update (wave-local; c/c_target live in VGPRs)
        float hnew[8];
        #pragma unroll
        for (int r = 0; r < 8; ++r) {
            float ig   = fast_sigm(acc[0][r]);
            float fg   = fast_sigm(acc[1][r]);
            float og   = fast_sigm(acc[2][r]);
            float itg  = fast_sigm(acc[3][r]);
            float ftg  = fast_sigm(acc[4][r]);
            float z    = fast_tanh(acc[5][r]);
            float cell = fg * cc[r] + ig * z;
            float h_i  = og * fast_tanh(cell);
            float ctn  = ftg * ctg[r] + itg * z;
            float cnew = ctn + (cell - ctn) * edtr[r];
            float hn   = og * fast_tanh(cnew);
            cc[r]  = cnew;
            ctg[r] = ctn;
            hnew[r] = hn;
            size_t o = ((size_t)t * B_SZ + b0 + r + 8 * hi) * H_DIM + wv * 16 + lm;
            __builtin_nontemporal_store(h_i,  out + o);              // hiddens
            __builtin_nontemporal_store(hn,   out + LBH + o);        // hiddens_ti
            __builtin_nontemporal_store(cell, out + 2 * LBH + o);    // cells
        }
        // decays [L,B,1]: lane 0 stores rows 0..7, lane 16 rows 8..15 (wave 0 only)
        if (wv == 0 && lm == 0) {
            #pragma unroll
            for (int r = 0; r < 8; ++r)
                __builtin_nontemporal_store(dec[r],
                    out + 3 * LBH + (size_t)t * B_SZ + b0 + r + 8 * hi);
        }

        // write h for step t+1 into ping-pong buffer; single barrier per step
        #pragma unroll
        for (int r = 0; r < 8; ++r)
            hnxt[(r + 8 * hi) * H_DIM + wv * 16 + lm] = (_Float16)hnew[r];
        __syncthreads();
    }
}

extern "C" void kernel_launch(void* const* d_in, const int* in_sizes, int n_in,
                              void* d_out, int out_size, void* d_ws, size_t ws_size,
                              hipStream_t stream) {
    const float* dt   = (const float*)d_in[0];
    const int*   ty   = (const int*)d_in[1];
    const float* h0   = (const float*)d_in[2];
    const float* c0   = (const float*)d_in[3];
    const float* emb  = (const float*)d_in[4];
    const float* Wi   = (const float*)d_in[5];  const float* bi  = (const float*)d_in[6];
    const float* Wf   = (const float*)d_in[7];  const float* bf  = (const float*)d_in[8];
    const float* Wo   = (const float*)d_in[9];  const float* bo  = (const float*)d_in[10];
    const float* Wit  = (const float*)d_in[11]; const float* bit = (const float*)d_in[12];
    const float* Wft  = (const float*)d_in[13]; const float* bft = (const float*)d_in[14];
    const float* Wz   = (const float*)d_in[15]; const float* bz  = (const float*)d_in[16];
    const float* Wd   = (const float*)d_in[17]; const float* bd  = (const float*)d_in[18];
    float* out = (float*)d_out;

    // Allow >48KB dynamic LDS (CDNA5 WGP supports up to 320KB per workgroup).
    (void)hipFuncSetAttribute((const void*)ctlstm_kernel,
                              hipFuncAttributeMaxDynamicSharedMemorySize, SMEM_BYTES);

    ctlstm_kernel<<<dim3(B_SZ / 16), dim3(256), SMEM_BYTES, stream>>>(
        dt, ty, h0, c0, emb, Wi, bi, Wf, bf, Wo, bo, Wit, bit, Wft, bft, Wz, bz,
        Wd, bd, out);
}
